// SpatioTemporalGNN_57466662420932
// MI455X (gfx1250) — compile-verified, matrix operand order checked by reading, not run on hardware
//
#include <hip/hip_runtime.h>
#include <math.h>

// ---------------- problem constants ----------------
#define NN 20000
#define EE 320000
#define TT 12
#define FF 7
#define HH 256
#define G3H 768
#define OUTD 12
#define LDHP 264            // bf16 LDS row stride (256 + 8 pad)

typedef __attribute__((ext_vector_type(16))) __bf16        v16bf;
typedef __attribute__((ext_vector_type(8)))  float         v8f;
typedef __attribute__((ext_vector_type(4)))  unsigned int  v4u;
typedef union { v4u q[2]; v16bf v; } fragu;

static __device__ inline v8f v8zero() {
  v8f z = {0.f,0.f,0.f,0.f,0.f,0.f,0.f,0.f};
  return z;
}
static __device__ inline unsigned short f2bf(float f) {
  unsigned int u = __float_as_uint(f);
  unsigned int r = u + 0x7fffu + ((u >> 16) & 1u);
  return (unsigned short)(r >> 16);
}
static __device__ inline float bf2f(unsigned short s) {
  return __uint_as_float(((unsigned int)s) << 16);
}
static __device__ inline float sigm(float v) { return 1.f / (1.f + __expf(-v)); }

// monotonic float<->uint key for atomicMax on signed floats
static __device__ inline unsigned int fenc(float f) {
  unsigned int u = __float_as_uint(f);
  return (u & 0x80000000u) ? ~u : (u | 0x80000000u);
}
static __device__ inline float fdec(unsigned int k) {
  return (k & 0x80000000u) ? __uint_as_float(k ^ 0x80000000u) : __uint_as_float(~k);
}

static __device__ inline v8f wmma_bf16(v16bf a, v16bf b, v8f c) {
  return __builtin_amdgcn_wmma_f32_16x16x32_bf16(false, a, false, b, (short)0, c, false, false);
}

// A fragment at a per-lane LDS pointer: two ds_load_b128 (offsets 0 / +32B)
static __device__ inline v16bf lda(const unsigned short* p) {
  fragu a;
  a.q[0] = *(const v4u*)(p);
  a.q[1] = *(const v4u*)(p + 16);
  return a.v;
}
// B fragment at a per-lane global pointer: two global_load_b128
static __device__ inline v16bf ldb(const unsigned int* p) {
  fragu b;
  b.q[0] = *(const v4u*)(p);
  b.q[1] = *(const v4u*)(p + 4);
  return b.v;
}

// ---------------- weight packing: fp32 [Nout,K] -> bf16 WMMA B fragments ----------------
// frag (kt, nt): dword slot = fragIdx*256 + lane*8 + i
// k(i,lane) = kt*32 + (i<4 ? 2i : 16+2(i-4)) + (lane>=16 ? 8 : 0); n = nt*16 + (lane&15)
__global__ void pack_bf16(const float* __restrict__ W, unsigned int* __restrict__ out,
                          int Nout, int NT, int K) {
  int tid = blockIdx.x * 256 + threadIdx.x;
  int KT = K >> 5;
  int total = KT * NT * 256;
  if (tid >= total) return;
  int f = tid >> 8;
  int r = tid & 255;
  int lane = r >> 3;
  int i = r & 7;
  int nt = f % NT;
  int kt = f / NT;
  int n = nt * 16 + (lane & 15);
  int k = kt * 32 + (i < 4 ? 2 * i : 16 + 2 * (i - 4)) + ((lane >> 4) << 3);
  unsigned int lo = 0, hi = 0;
  if (n < Nout) {
    lo = f2bf(W[(size_t)n * K + k]);
    hi = f2bf(W[(size_t)n * K + k + 1]);
  }
  out[tid] = lo | (hi << 16);
}

// ---------------- generic WMMA GEMM (K=256): C[M,Nc] = act(A @ Bpacked + bias) ----------------
// Each wave: n-tiles {wave, wave+8}, 4 independent accumulator chains
// (2 tiles x even/odd-kt split-K) to hide WMMA->WMMA hazards. Pointer-increment
// addressing + unroll(1) keeps the fragment live range to one batch (no spills).
template <int NTc>
__global__ __launch_bounds__(256) void gemm_t(const float* __restrict__ A,
                                              const unsigned int* __restrict__ Bp,
                                              const float* __restrict__ bias,
                                              float* __restrict__ Cout,
                                              int M, int Nc, int ldc, int act) {
  __shared__ __align__(16) unsigned short Alds[16 * LDHP];
  const int m0 = blockIdx.x * 16;
  for (int idx = threadIdx.x; idx < 16 * HH; idx += 256) {
    int r = idx >> 8, c = idx & 255;
    float v = (m0 + r < M) ? A[(size_t)(m0 + r) * HH + c] : 0.f;
    Alds[r * LDHP + c] = f2bf(v);
  }
  __syncthreads();
  const int wave = threadIdx.x >> 5;
  const int lane = threadIdx.x & 31;
  const int row16 = lane & 15;
  const int khalf = lane >> 4;
  const int nt0 = wave;
  const int nt1 = wave + 8;
  constexpr bool H1 = (NTc > 8);
  if (nt0 >= NTc) return;

  const unsigned int* p0 = Bp + (nt0 << 8) + (lane << 3);
  const unsigned int* p1 = Bp + (nt1 << 8) + (lane << 3);
  __builtin_prefetch(p0, 0, 3);
  if (H1) __builtin_prefetch(p1, 0, 3);
  const unsigned short* ap = Alds + row16 * LDHP + khalf * 8;

  v8f aA0 = v8zero(), aB0 = v8zero(), aA1 = v8zero(), aB1 = v8zero();
#pragma unroll 1
  for (int kt = 0; kt < 8; kt += 2) {
    v16bf ae = lda(ap);
    v16bf ao = lda(ap + 32);
    v16bf be0 = ldb(p0);
    v16bf bo0 = ldb(p0 + (NTc << 8));
    if constexpr (H1) {
      v16bf be1 = ldb(p1);
      v16bf bo1 = ldb(p1 + (NTc << 8));
      aA0 = wmma_bf16(ae, be0, aA0);
      aA1 = wmma_bf16(ae, be1, aA1);
      aB0 = wmma_bf16(ao, bo0, aB0);
      aB1 = wmma_bf16(ao, bo1, aB1);
    } else {
      aA0 = wmma_bf16(ae, be0, aA0);
      aB0 = wmma_bf16(ao, bo0, aB0);
    }
    ap += 64;
    p0 += 2 * (NTc << 8);
    p1 += 2 * (NTc << 8);
  }
  {
    v8f acc = aA0 + aB0;
    int col = nt0 * 16 + (lane & 15);
    if (col < Nc) {
      float bv = bias[col];
      const float* pa = (const float*)&acc;
#pragma unroll
      for (int i = 0; i < 8; ++i) {
        int gr = m0 + i + (khalf << 3);
        if (gr < M) {
          float v = pa[i] + bv;
          if (act) v = v > 0.f ? v : 0.f;
          Cout[(size_t)gr * ldc + col] = v;
        }
      }
    }
  }
  if constexpr (H1) {
    v8f acc = aA1 + aB1;
    int col = nt1 * 16 + (lane & 15);
    if (col < Nc) {
      float bv = bias[col];
      const float* pa = (const float*)&acc;
#pragma unroll
      for (int i = 0; i < 8; ++i) {
        int gr = m0 + i + (khalf << 3);
        if (gr < M) {
          float v = pa[i] + bv;
          if (act) v = v > 0.f ? v : 0.f;
          Cout[(size_t)gr * ldc + col] = v;
        }
      }
    }
  }
}

// ---------------- fused 2-layer GRU over T steps (per-node independent) ----------------
__global__ __launch_bounds__(256) void gru_kernel(
    const float* __restrict__ x, const float* __restrict__ Wih0,
    const float* __restrict__ bih0, const float* __restrict__ bhh0,
    const unsigned int* __restrict__ Bhh0, const unsigned int* __restrict__ Bih1,
    const unsigned int* __restrict__ Bhh1,
    const float* __restrict__ bih1, const float* __restrict__ bhh1,
    float* __restrict__ Hout) {
  __shared__ __align__(16) float xls[16 * 88];
  __shared__ __align__(16) unsigned short h0b[16 * LDHP];
  __shared__ __align__(16) unsigned short h1b[16 * LDHP];
  const int m0 = blockIdx.x * 16;
  for (int idx = threadIdx.x; idx < 16 * (TT * FF); idx += 256) {
    int r = idx / (TT * FF), c = idx % (TT * FF);
    xls[r * 88 + c] = x[(size_t)(m0 + r) * (TT * FF) + c];
  }
  for (int idx = threadIdx.x; idx < 16 * LDHP; idx += 256) { h0b[idx] = 0; h1b[idx] = 0; }

  const int wave = threadIdx.x >> 5;
  const int lane = threadIdx.x & 31;
  const int c0 = wave * 32;
  const int row16 = lane & 15;
  const int khalf = lane >> 4;
  const int rbase = khalf << 3;
  const int lcol = lane & 15;

  // per-lane B base pointers (column pair ntb, ntb+1 via +256-dword immediates)
  const unsigned int* baseHH0 = Bhh0 + ((wave * 2) << 8) + (lane << 3);
  const unsigned int* baseIH1 = Bih1 + ((wave * 2) << 8) + (lane << 3);
  const unsigned int* baseHH1 = Bhh1 + ((wave * 2) << 8) + (lane << 3);
  __builtin_prefetch(baseHH0, 0, 3);
  __builtin_prefetch(baseIH1, 0, 3);
  __builtin_prefetch(baseHH1, 0, 3);
  const unsigned short* a0base = h0b + row16 * LDHP + khalf * 8;
  const unsigned short* a1base = h1b + row16 * LDHP + khalf * 8;

  // hoist per-lane input weights (K=7) and fused biases
  float wih[2][3][7];
  float bR0[2], bZ0[2], bIN0[2], bHN0[2];
  float bR1[2], bZ1[2], bIN1[2], bHN1[2];
  for (int hlf = 0; hlf < 2; ++hlf) {
    int col = c0 + hlf * 16 + lcol;
    for (int f = 0; f < 7; ++f) {
      wih[hlf][0][f] = Wih0[(size_t)col * 7 + f];
      wih[hlf][1][f] = Wih0[(size_t)(256 + col) * 7 + f];
      wih[hlf][2][f] = Wih0[(size_t)(512 + col) * 7 + f];
    }
    bR0[hlf] = bih0[col] + bhh0[col];
    bZ0[hlf] = bih0[256 + col] + bhh0[256 + col];
    bIN0[hlf] = bih0[512 + col];
    bHN0[hlf] = bhh0[512 + col];
    bR1[hlf] = bih1[col] + bhh1[col];
    bZ1[hlf] = bih1[256 + col] + bhh1[256 + col];
    bIN1[hlf] = bih1[512 + col];
    bHN1[hlf] = bhh1[512 + col];
  }
  __syncthreads();

#pragma unroll 1
  for (int t = 0; t < TT; ++t) {
    // ===================== layer 0 =====================
    v8f aR[2], aZ[2], aIN[2], aHN[2];
#pragma unroll
    for (int hlf = 0; hlf < 2; ++hlf) { aR[hlf] = v8zero(); aZ[hlf] = v8zero(); aIN[hlf] = v8zero(); aHN[hlf] = v8zero(); }
    // gi = x_t @ Wih0^T (K=7 -> VALU, written straight into the WMMA C layout)
#pragma unroll
    for (int i = 0; i < 8; ++i) {
      int r = i + rbase;
      float xv[7];
#pragma unroll
      for (int f = 0; f < 7; ++f) xv[f] = xls[r * 88 + t * 7 + f];
#pragma unroll
      for (int hlf = 0; hlf < 2; ++hlf) {
        float s0 = 0.f, s1 = 0.f, s2 = 0.f;
#pragma unroll
        for (int f = 0; f < 7; ++f) {
          s0 += xv[f] * wih[hlf][0][f];
          s1 += xv[f] * wih[hlf][1][f];
          s2 += xv[f] * wih[hlf][2][f];
        }
        ((float*)&aR[hlf])[i] = s0;
        ((float*)&aZ[hlf])[i] = s1;
        ((float*)&aIN[hlf])[i] = s2;
      }
    }
    // gh = h0 @ Whh0^T : per kt, 1 A frag + 6 B frags then 6 independent WMMAs
    {
      const unsigned short* ap = a0base;
      const unsigned int* p = baseHH0;
#pragma unroll 1
      for (int kt = 0; kt < 8; ++kt) {
        v16bf a = lda(ap);
        v16bf br0 = ldb(p);
        v16bf br1 = ldb(p + (1 << 8));
        v16bf bz0 = ldb(p + (16 << 8));
        v16bf bz1 = ldb(p + (17 << 8));
        v16bf bn0 = ldb(p + (32 << 8));
        v16bf bn1 = ldb(p + (33 << 8));
        aR[0] = wmma_bf16(a, br0, aR[0]);
        aR[1] = wmma_bf16(a, br1, aR[1]);
        aZ[0] = wmma_bf16(a, bz0, aZ[0]);
        aZ[1] = wmma_bf16(a, bz1, aZ[1]);
        aHN[0] = wmma_bf16(a, bn0, aHN[0]);
        aHN[1] = wmma_bf16(a, bn1, aHN[1]);
        ap += 32;
        p += (48 << 8);
      }
    }
    float hnew0[2][8];
#pragma unroll
    for (int hlf = 0; hlf < 2; ++hlf) {
      int col = c0 + hlf * 16 + lcol;
      const float* pr = (const float*)&aR[hlf];
      const float* pz = (const float*)&aZ[hlf];
      const float* pi = (const float*)&aIN[hlf];
      const float* ph = (const float*)&aHN[hlf];
#pragma unroll
      for (int i = 0; i < 8; ++i) {
        float rg = sigm(pr[i] + bR0[hlf]);
        float zg = sigm(pz[i] + bZ0[hlf]);
        float ng = tanhf(pi[i] + bIN0[hlf] + rg * (ph[i] + bHN0[hlf]));
        float hp = bf2f(h0b[(i + rbase) * LDHP + col]);
        hnew0[hlf][i] = (1.f - zg) * ng + zg * hp;
      }
    }
    __syncthreads();
#pragma unroll
    for (int hlf = 0; hlf < 2; ++hlf) {
      int col = c0 + hlf * 16 + lcol;
#pragma unroll
      for (int i = 0; i < 8; ++i) h0b[(i + rbase) * LDHP + col] = f2bf(hnew0[hlf][i]);
    }
    __syncthreads();

    // ===================== layer 1 =====================
#pragma unroll
    for (int hlf = 0; hlf < 2; ++hlf) { aR[hlf] = v8zero(); aZ[hlf] = v8zero(); aIN[hlf] = v8zero(); aHN[hlf] = v8zero(); }
    // gi1: A = h0 (this step's layer-0 output)
    {
      const unsigned short* ap = a0base;
      const unsigned int* p = baseIH1;
#pragma unroll 1
      for (int kt = 0; kt < 8; ++kt) {
        v16bf a0 = lda(ap);
        v16bf br0 = ldb(p);
        v16bf br1 = ldb(p + (1 << 8));
        v16bf bz0 = ldb(p + (16 << 8));
        v16bf bz1 = ldb(p + (17 << 8));
        v16bf bn0 = ldb(p + (32 << 8));
        v16bf bn1 = ldb(p + (33 << 8));
        aR[0] = wmma_bf16(a0, br0, aR[0]);
        aR[1] = wmma_bf16(a0, br1, aR[1]);
        aZ[0] = wmma_bf16(a0, bz0, aZ[0]);
        aZ[1] = wmma_bf16(a0, bz1, aZ[1]);
        aIN[0] = wmma_bf16(a0, bn0, aIN[0]);
        aIN[1] = wmma_bf16(a0, bn1, aIN[1]);
        ap += 32;
        p += (48 << 8);
      }
    }
    // gh1: A = h1 (previous hidden)
    {
      const unsigned short* ap = a1base;
      const unsigned int* q = baseHH1;
#pragma unroll 1
      for (int kt = 0; kt < 8; ++kt) {
        v16bf a1 = lda(ap);
        v16bf br0 = ldb(q);
        v16bf br1 = ldb(q + (1 << 8));
        v16bf bz0 = ldb(q + (16 << 8));
        v16bf bz1 = ldb(q + (17 << 8));
        v16bf bn0 = ldb(q + (32 << 8));
        v16bf bn1 = ldb(q + (33 << 8));
        aR[0] = wmma_bf16(a1, br0, aR[0]);
        aR[1] = wmma_bf16(a1, br1, aR[1]);
        aZ[0] = wmma_bf16(a1, bz0, aZ[0]);
        aZ[1] = wmma_bf16(a1, bz1, aZ[1]);
        aHN[0] = wmma_bf16(a1, bn0, aHN[0]);
        aHN[1] = wmma_bf16(a1, bn1, aHN[1]);
        ap += 32;
        q += (48 << 8);
      }
    }
    float hnew1[2][8];
#pragma unroll
    for (int hlf = 0; hlf < 2; ++hlf) {
      int col = c0 + hlf * 16 + lcol;
      const float* pr = (const float*)&aR[hlf];
      const float* pz = (const float*)&aZ[hlf];
      const float* pi = (const float*)&aIN[hlf];
      const float* ph = (const float*)&aHN[hlf];
#pragma unroll
      for (int i = 0; i < 8; ++i) {
        float rg = sigm(pr[i] + bR1[hlf]);
        float zg = sigm(pz[i] + bZ1[hlf]);
        float ng = tanhf(pi[i] + bIN1[hlf] + rg * (ph[i] + bHN1[hlf]));
        float hp = bf2f(h1b[(i + rbase) * LDHP + col]);
        hnew1[hlf][i] = (1.f - zg) * ng + zg * hp;
      }
    }
    __syncthreads();
#pragma unroll
    for (int hlf = 0; hlf < 2; ++hlf) {
      int col = c0 + hlf * 16 + lcol;
#pragma unroll
      for (int i = 0; i < 8; ++i) h1b[(i + rbase) * LDHP + col] = f2bf(hnew1[hlf][i]);
    }
    __syncthreads();
  }
  for (int idx = threadIdx.x; idx < 16 * HH; idx += 256) {
    int r = idx >> 8, c = idx & 255;
    Hout[(size_t)(m0 + r) * HH + c] = bf2f(h1b[r * LDHP + c]);
  }
}

// ---------------- GATv2 edge kernels ----------------
__global__ void edge_alpha(const float* __restrict__ xl, const float* __restrict__ xr,
                           const float* __restrict__ ea, const float* __restrict__ Wev,
                           const float* __restrict__ attv, const int* __restrict__ src,
                           const int* __restrict__ dst, float* __restrict__ alpha,
                           unsigned int* __restrict__ menc, int Ecnt) {
  int e = blockIdx.x * 256 + threadIdx.x;
  if (e >= Ecnt) return;
  int s = src[e], d = dst[e];
  float eav = ea[e];
  const float4* xj = (const float4*)(xl + (size_t)s * HH);
  const float4* xi = (const float4*)(xr + (size_t)d * HH);
  const float4* we = (const float4*)Wev;
  const float4* at = (const float4*)attv;
  float acc[4] = {0.f, 0.f, 0.f, 0.f};
  for (int q = 0; q < 64; ++q) {
    float4 a = xi[q], b = xj[q], w = we[q], av = at[q];
    int h = q >> 4;
    float s0 = a.x + b.x + eav * w.x; s0 = s0 > 0.f ? s0 : 0.2f * s0;
    float s1 = a.y + b.y + eav * w.y; s1 = s1 > 0.f ? s1 : 0.2f * s1;
    float s2 = a.z + b.z + eav * w.z; s2 = s2 > 0.f ? s2 : 0.2f * s2;
    float s3 = a.w + b.w + eav * w.w; s3 = s3 > 0.f ? s3 : 0.2f * s3;
    acc[h] += s0 * av.x + s1 * av.y + s2 * av.z + s3 * av.w;
  }
  for (int h = 0; h < 4; ++h) {
    alpha[(size_t)e * 4 + h] = acc[h];
    atomicMax(&menc[(size_t)d * 4 + h], fenc(acc[h]));
  }
}

__global__ void edge_expsum(float* __restrict__ alpha, const unsigned int* __restrict__ menc,
                            float* __restrict__ den, const int* __restrict__ dst, int Ecnt) {
  int e = blockIdx.x * 256 + threadIdx.x;
  if (e >= Ecnt) return;
  int d = dst[e];
  for (int h = 0; h < 4; ++h) {
    float m = fdec(menc[(size_t)d * 4 + h]);
    float a = __expf(alpha[(size_t)e * 4 + h] - m);
    alpha[(size_t)e * 4 + h] = a;
    atomicAdd(&den[(size_t)d * 4 + h], a);
  }
}

__global__ void edge_scatter(const float* __restrict__ alpha, const float* __restrict__ den,
                             const float* __restrict__ xl, const int* __restrict__ src,
                             const int* __restrict__ dst, float* __restrict__ msg, int Ecnt) {
  int e = blockIdx.x * 256 + threadIdx.x;
  if (e >= Ecnt) return;
  int s = src[e], d = dst[e];
  float w[4];
  for (int h = 0; h < 4; ++h) w[h] = alpha[(size_t)e * 4 + h] / den[(size_t)d * 4 + h];
  const float4* xj = (const float4*)(xl + (size_t)s * HH);
  float* o = msg + (size_t)d * HH;
  for (int q = 0; q < 64; ++q) {
    float4 v = xj[q];
    float wh = w[q >> 4];
    atomicAdd(&o[q * 4 + 0], v.x * wh);
    atomicAdd(&o[q * 4 + 1], v.y * wh);
    atomicAdd(&o[q * 4 + 2], v.z * wh);
    atomicAdd(&o[q * 4 + 3], v.w * wh);
  }
}

// ---------------- LayerNorm + ELU + residual (one wave32 per node) ----------------
__global__ __launch_bounds__(256) void ln_elu_res(const float* __restrict__ msg,
                                                  const float* __restrict__ cb,
                                                  const float* __restrict__ g,
                                                  const float* __restrict__ b,
                                                  float* __restrict__ h) {
  int node = blockIdx.x * 8 + (threadIdx.x >> 5);
  int lane = threadIdx.x & 31;
  if (node >= NN) return;
  const float* m = msg + (size_t)node * HH;
  float* hp = h + (size_t)node * HH;
  float v[8];
  float sum = 0.f;
  for (int i = 0; i < 8; ++i) {
    int c = lane + 32 * i;
    v[i] = m[c] + cb[c];
    sum += v[i];
  }
  for (int off = 16; off > 0; off >>= 1) sum += __shfl_xor(sum, off, 32);
  float mu = sum * (1.f / HH);
  float vs = 0.f;
  for (int i = 0; i < 8; ++i) { float d = v[i] - mu; vs += d * d; }
  for (int off = 16; off > 0; off >>= 1) vs += __shfl_xor(vs, off, 32);
  float inv = rsqrtf(vs * (1.f / HH) + 1e-5f);
  for (int i = 0; i < 8; ++i) {
    int c = lane + 32 * i;
    float z = (v[i] - mu) * inv * g[c] + b[c];
    z = z > 0.f ? z : (__expf(z) - 1.f);
    hp[c] = z + hp[c];
  }
}

__global__ void zero_u32(unsigned int* p, int n) {
  int i = blockIdx.x * 256 + threadIdx.x;
  if (i < n) p[i] = 0;
}

// ---------------- host orchestration ----------------
extern "C" void kernel_launch(void* const* d_in, const int* in_sizes, int n_in,
                              void* d_out, int out_size, void* d_ws, size_t ws_size,
                              hipStream_t stream) {
  (void)in_sizes; (void)n_in; (void)out_size; (void)ws_size;
  const float* x     = (const float*)d_in[0];
  const float* eattr = (const float*)d_in[1];
  const float* Wih0  = (const float*)d_in[2];
  const float* Whh0  = (const float*)d_in[3];
  const float* bih0  = (const float*)d_in[4];
  const float* bhh0  = (const float*)d_in[5];
  const float* Wih1  = (const float*)d_in[6];
  const float* Whh1  = (const float*)d_in[7];
  const float* bih1  = (const float*)d_in[8];
  const float* bhh1  = (const float*)d_in[9];
  const float* Wl    = (const float*)d_in[10];
  const float* bl    = (const float*)d_in[11];
  const float* Wr    = (const float*)d_in[12];
  const float* br    = (const float*)d_in[13];
  const float* We    = (const float*)d_in[14];
  const float* att   = (const float*)d_in[15];
  const float* convb = (const float*)d_in[16];
  const float* lng   = (const float*)d_in[17];
  const float* lnb   = (const float*)d_in[18];
  const float* fcW1  = (const float*)d_in[19];
  const float* fcb1  = (const float*)d_in[20];
  const float* fcW2  = (const float*)d_in[21];
  const float* fcb2  = (const float*)d_in[22];
  const int*   ei    = (const int*)d_in[23];
  const int* srcI = ei;
  const int* dstI = ei + EE;

  char* ws = (char*)d_ws;
  size_t off = 0;
  auto alloc = [&](size_t bytes) -> void* {
    void* p = ws + off;
    off = (off + bytes + 255) & ~(size_t)255;
    return p;
  };
  const size_t szWhh  = (size_t)8 * 48 * 256 * 4;  // 768x256 packed
  const size_t szW256 = (size_t)8 * 16 * 256 * 4;  // 256x256 packed
  unsigned int* pWhh0 = (unsigned int*)alloc(szWhh);
  unsigned int* pWih1 = (unsigned int*)alloc(szWhh);
  unsigned int* pWhh1 = (unsigned int*)alloc(szWhh);
  unsigned int* pWl[3], *pWr[3];
  for (int l = 0; l < 3; ++l) pWl[l] = (unsigned int*)alloc(szW256);
  for (int l = 0; l < 3; ++l) pWr[l] = (unsigned int*)alloc(szW256);
  unsigned int* pF1 = (unsigned int*)alloc(szW256);
  unsigned int* pF2 = (unsigned int*)alloc((size_t)8 * 1 * 256 * 4);
  float* H   = (float*)alloc((size_t)NN * HH * 4);
  float* XL  = (float*)alloc((size_t)NN * HH * 4);
  float* XR  = (float*)alloc((size_t)NN * HH * 4);
  float* MSG = (float*)alloc((size_t)NN * HH * 4);
  float* ALPHA = (float*)alloc((size_t)EE * 4 * 4);
  unsigned int* MENC = (unsigned int*)alloc((size_t)NN * 4 * 4);
  float* DEN = (float*)alloc((size_t)NN * 4 * 4);

  // pack all GEMM weights to bf16 WMMA fragments
  pack_bf16<<<(8 * 48 * 256 + 255) / 256, 256, 0, stream>>>(Whh0, pWhh0, G3H, 48, HH);
  pack_bf16<<<(8 * 48 * 256 + 255) / 256, 256, 0, stream>>>(Wih1, pWih1, G3H, 48, HH);
  pack_bf16<<<(8 * 48 * 256 + 255) / 256, 256, 0, stream>>>(Whh1, pWhh1, G3H, 48, HH);
  for (int l = 0; l < 3; ++l) {
    pack_bf16<<<(8 * 16 * 256 + 255) / 256, 256, 0, stream>>>(Wl + (size_t)l * HH * HH, pWl[l], HH, 16, HH);
    pack_bf16<<<(8 * 16 * 256 + 255) / 256, 256, 0, stream>>>(Wr + (size_t)l * HH * HH, pWr[l], HH, 16, HH);
  }
  pack_bf16<<<(8 * 16 * 256 + 255) / 256, 256, 0, stream>>>(fcW1, pF1, HH, 16, HH);
  pack_bf16<<<(8 * 1 * 256 + 255) / 256, 256, 0, stream>>>(fcW2, pF2, OUTD, 1, HH);

  // temporal encoder: fused 2-layer GRU -> H [N,256]
  gru_kernel<<<NN / 16, 256, 0, stream>>>(x, Wih0, bih0, bhh0, pWhh0, pWih1, pWhh1, bih1, bhh1, H);

  // stacked GATv2 blocks
  for (int l = 0; l < 3; ++l) {
    zero_u32<<<(NN * HH + 255) / 256, 256, 0, stream>>>((unsigned int*)MSG, NN * HH);
    zero_u32<<<(NN * 4 + 255) / 256, 256, 0, stream>>>(MENC, NN * 4);
    zero_u32<<<(NN * 4 + 255) / 256, 256, 0, stream>>>((unsigned int*)DEN, NN * 4);
    gemm_t<16><<<NN / 16, 256, 0, stream>>>(H, pWl[l], bl + l * HH, XL, NN, HH, HH, 0);
    gemm_t<16><<<NN / 16, 256, 0, stream>>>(H, pWr[l], br + l * HH, XR, NN, HH, HH, 0);
    edge_alpha<<<(EE + 255) / 256, 256, 0, stream>>>(XL, XR, eattr, We + l * HH, att + l * HH,
                                                     srcI, dstI, ALPHA, MENC, EE);
    edge_expsum<<<(EE + 255) / 256, 256, 0, stream>>>(ALPHA, MENC, DEN, dstI, EE);
    edge_scatter<<<(EE + 255) / 256, 256, 0, stream>>>(ALPHA, DEN, XL, srcI, dstI, MSG, EE);
    ln_elu_res<<<(NN + 7) / 8, 256, 0, stream>>>(MSG, convb + l * HH, lng + l * HH, lnb + l * HH, H);
  }

  // prediction head
  gemm_t<16><<<NN / 16, 256, 0, stream>>>(H, pF1, fcb1, XL, NN, HH, HH, 1);
  gemm_t<1><<<NN / 16, 256, 0, stream>>>(XL, pF2, fcb2, (float*)d_out, NN, OUTD, OUTD, 0);
}
